// AutoformerAttention_55327768707475
// MI455X (gfx1250) — matrix operand order repaired and checked by reading, not compiled, not run on hardware
//
#include <hip/hip_runtime.h>
#include <hip/hip_bf16.h>
#include <cstdint>

// ===================== MI455X / gfx1250 implementation =====================
// Pipeline:
//   1) convert X, Wq/Wk/Wv/Wo to bf16
//   2) Q,K,V = bf16 WMMA GEMMs (X @ W^T + b), f32 accum, TDM tile loads to LDS
//   3) per-(b, channel-group) length-2048 LDS FFT of z=q+ik, accumulate
//      sum_c Qf*conj(Kf) into per-group partial spectra (deterministic)
//   4) reduce partials + inverse FFT -> corr_mean[b, tau]
//   5) top-22 + softmax per batch
//   6) weighted circular-shift aggregation of V (bandwidth bound, float4)
//   7) out = bf16 WMMA GEMM (agg @ Wo^T + bo) -> d_out (fp32)

typedef __bf16 v16bf __attribute__((ext_vector_type(16)));
typedef __bf16 v8bf  __attribute__((ext_vector_type(8)));
typedef float  v8f   __attribute__((ext_vector_type(8)));
typedef unsigned int u32;
typedef u32  u32x4 __attribute__((ext_vector_type(4)));
typedef int  i32x8 __attribute__((ext_vector_type(8)));
typedef int  i32x4 __attribute__((ext_vector_type(4)));

#define BB   16
#define TT   2048
#define EEC  1024
#define TOPK 22

#define BM 128
#define BN 64
#define BK 32

#define NEG_INF (-3.402823466e+38f)

// ---------------------------------------------------------------------------
// TDM: 2D tile load (global -> LDS), D# built per CDNA5 ISA ch.8.
// data_size = 2 bytes (bf16). Issued per-wave; callers gate to wave 0.
// 6-arg builtin form (clang-23 / therock-10.0 headers).
// ---------------------------------------------------------------------------
__device__ __forceinline__ void tdm_load_2d_bf16(u32 lds_byte_off, const void* gtile,
                                                 u32 tile_w, u32 tile_h,
                                                 u32 tensor_w, u32 tensor_h,
                                                 u32 row_stride_elems) {
  unsigned long long ga = (unsigned long long)(uintptr_t)gtile;
  u32x4 g0;
  g0[0] = 1u;                                             // count=1 (valid user D#)
  g0[1] = lds_byte_off;                                   // lds_addr
  g0[2] = (u32)ga;                                        // global_addr[31:0]
  g0[3] = (u32)((ga >> 32) & 0x01FFFFFFu) | (2u << 30);   // global_addr[56:32], type=2
  i32x8 g1;
  g1[0] = (int)(1u << 16);                                // data_size=1 -> 2 bytes
  g1[1] = (int)((tensor_w & 0xFFFFu) << 16);              // tensor_dim0[15:0]
  g1[2] = (int)((tensor_w >> 16) | ((tensor_h & 0xFFFFu) << 16));
  g1[3] = (int)((tensor_h >> 16) | (tile_w << 16));       // tile_dim0
  g1[4] = (int)tile_h;                                    // tile_dim1 (tile_dim2=0)
  g1[5] = (int)row_stride_elems;                          // tensor_dim0_stride[31:0]
  g1[6] = 0;
  g1[7] = 0;
  i32x4 gz4; gz4[0] = 0; gz4[1] = 0; gz4[2] = 0; gz4[3] = 0;
  i32x8 gz8;
  gz8[0] = 0; gz8[1] = 0; gz8[2] = 0; gz8[3] = 0;
  gz8[4] = 0; gz8[5] = 0; gz8[6] = 0; gz8[7] = 0;
  __builtin_amdgcn_tensor_load_to_lds(g0, g1, gz4, gz4, gz8, 0);
}

__device__ __forceinline__ u32 lds_off_of(const void* p) {
  // gfx1250 flat->LDS mapping uses addr[31:0] as the LDS byte address.
  return (u32)(uintptr_t)p;
}

// ---------------------------------------------------------------------------
// fp32 -> bf16 conversion
// ---------------------------------------------------------------------------
__global__ __launch_bounds__(256) void convert_f32_to_bf16(
    const float* __restrict__ src, __bf16* __restrict__ dst, long long n) {
  long long i = (long long)blockIdx.x * 256 + threadIdx.x;
  if (i < n) dst[i] = (__bf16)src[i];
}

// ---------------------------------------------------------------------------
// bf16 WMMA GEMM:  C[m,n] = sum_k A[m,k]*W[n,k] + bias[n]   (f32 accumulate)
// Block 256 thr = 8 waves; tile BM x BN; waves 4(M) x 2(N), 32x32 per wave.
// A/W tiles staged into LDS by the Tensor Data Mover (wave 0 issues).
// ---------------------------------------------------------------------------
__global__ __launch_bounds__(256) void gemm_bf16_wmma(
    const __bf16* __restrict__ A, const __bf16* __restrict__ W,
    const float* __restrict__ bias, float* __restrict__ C,
    int M, int N, int K) {
  __shared__ __align__(16) __bf16 smem[BM * BK + BN * BK];
  __bf16* As = smem;                // [BM][BK] row-major
  __bf16* Bs = smem + BM * BK;      // [BN][BK] row-major

  const int tid  = threadIdx.x;
  const int lane = tid & 31;
  const int wave = tid >> 5;        // 0..7
  const int wm   = wave >> 1;       // 0..3  (M sub-tile)
  const int wn   = wave & 1;        // 0..1  (N sub-tile)
  const int mBase = blockIdx.y * BM;
  const int nBase = blockIdx.x * BN;

  const u32 ldsA = lds_off_of(As);
  const u32 ldsB = lds_off_of(Bs);

  const int hf  = lane >> 4;        // lane half (0/1)
  const int l16 = lane & 15;

  v8f acc[2][2] = {};

  for (int k0 = 0; k0 < K; k0 += BK) {
    __syncthreads();                               // prior tile fully consumed
    if (wave == 0) {
      tdm_load_2d_bf16(ldsA, &A[(size_t)mBase * K + k0], BK, BM, (u32)K, (u32)M, (u32)K);
      tdm_load_2d_bf16(ldsB, &W[(size_t)nBase * K + k0], BK, BN, (u32)K, (u32)N, (u32)K);
      __builtin_amdgcn_s_wait_tensorcnt(0);
    }
    __syncthreads();                               // tile visible to all waves

    // B fragments: lane holds column n=l16; K chunk = 16*hf .. 16*hf+15
    v16bf bfrag[2];
#pragma unroll
    for (int nt = 0; nt < 2; ++nt) {
      const int col = wn * 32 + nt * 16 + l16;
      const v8bf* bp = (const v8bf*)&Bs[col * BK];
      v8bf blo = bp[2 * hf];
      v8bf bhi = bp[2 * hf + 1];
      bfrag[nt] = __builtin_shufflevector(blo, bhi,
                    0,1,2,3,4,5,6,7,8,9,10,11,12,13,14,15);
    }
#pragma unroll
    for (int mt = 0; mt < 2; ++mt) {
      // A fragment: lane holds row m=l16; K chunk = 16*hf .. 16*hf+15
      const int row = wm * 32 + mt * 16 + l16;
      const v8bf* ap = (const v8bf*)&As[row * BK];
      v8bf alo = ap[2 * hf];
      v8bf ahi = ap[2 * hf + 1];
      v16bf afrag = __builtin_shufflevector(alo, ahi,
                      0,1,2,3,4,5,6,7,8,9,10,11,12,13,14,15);
#pragma unroll
      for (int nt = 0; nt < 2; ++nt) {
        acc[mt][nt] = __builtin_amdgcn_wmma_f32_16x16x32_bf16(
            false, afrag, false, bfrag[nt], (short)0, acc[mt][nt], false, false);
      }
    }
  }

  // Epilogue: D layout -> element (r, lane): m = r + 8*hf, n = l16
#pragma unroll
  for (int mt = 0; mt < 2; ++mt) {
#pragma unroll
    for (int nt = 0; nt < 2; ++nt) {
      const int n  = nBase + wn * 32 + nt * 16 + l16;
      const float bval = bias[n];
      const int m0 = mBase + wm * 32 + mt * 16 + (hf << 3);
#pragma unroll
      for (int r = 0; r < 8; ++r) {
        C[(size_t)(m0 + r) * N + n] = acc[mt][nt][r] + bval;
      }
    }
  }
}

// ---------------------------------------------------------------------------
// In-place radix-2 DIT FFT over LDS arrays (length TT), after bit-reversal.
// sign = -1 forward, +1 inverse (unnormalized).
// ---------------------------------------------------------------------------
__device__ __forceinline__ void block_fft(float* re, float* im, float sign) {
  for (int hl = 1; hl < TT; hl <<= 1) {
    const float ang = sign * 3.14159265358979323846f / (float)hl;
    for (int i = threadIdx.x; i < (TT >> 1); i += 256) {
      const int off = i & (hl - 1);
      const int p   = ((i & ~(hl - 1)) << 1) | off;
      const int q   = p + hl;
      float s, c;
      __sincosf(ang * (float)off, &s, &c);
      const float ur = re[p], ui = im[p];
      const float vr = re[q], vi = im[q];
      const float tr = vr * c - vi * s;
      const float ti = vr * s + vi * c;
      re[p] = ur + tr; im[p] = ui + ti;
      re[q] = ur - tr; im[q] = ui - ti;
    }
    __syncthreads();
  }
}

// ---------------------------------------------------------------------------
// Per-(b, 16-channel group): FFT(q + i*k) per channel, accumulate
// Qf*conj(Kf) into a group-partial spectrum. Deterministic (no atomics).
// part layout: [B][64][TT][2] float
// ---------------------------------------------------------------------------
__global__ __launch_bounds__(256) void fft_corr_partial(
    const float* __restrict__ Q, const float* __restrict__ Kx,
    float* __restrict__ part) {
  __shared__ float re[TT], im[TT];
  __shared__ float ar[TT], ai[TT];
  const int g = blockIdx.x;   // channel group 0..63
  const int b = blockIdx.y;

  for (int f = threadIdx.x; f < TT; f += 256) { ar[f] = 0.f; ai[f] = 0.f; }

  for (int cc = 0; cc < 16; ++cc) {
    const int c = g * 16 + cc;
    const size_t base = (size_t)b * TT * EEC + c;
    __syncthreads();                               // re/im free for reuse
    for (int j = threadIdx.x; j < TT; j += 256) {
      const int r = __brev((unsigned)j) >> 21;     // 11-bit reversal
      re[r] = Q[base + (size_t)j * EEC];
      im[r] = Kx[base + (size_t)j * EEC];
    }
    __syncthreads();
    block_fft(re, im, -1.0f);                      // forward
    // Z[f] = Qf + i*Kf : unpack and accumulate Qf * conj(Kf)
    for (int f = threadIdx.x; f < TT; f += 256) {
      const int fn = (TT - f) & (TT - 1);
      const float zr = re[f],  zi = im[f];
      const float wr = re[fn], wi = im[fn];
      const float qr = 0.5f * (zr + wr), qi = 0.5f * (zi - wi);
      const float dr = zr - wr,          di = zi + wi;
      const float kr = 0.5f * di,        ki = -0.5f * dr;
      ar[f] += qr * kr + qi * ki;                  // Re(Qf*conj(Kf))
      ai[f] += qi * kr - qr * ki;                  // Im(Qf*conj(Kf))
    }
  }
  __syncthreads();
  for (int f = threadIdx.x; f < TT; f += 256) {
    const size_t o = (((size_t)b * 64 + g) * TT + f) * 2;
    part[o]     = ar[f];
    part[o + 1] = ai[f];
  }
}

// ---------------------------------------------------------------------------
// Reduce 64 group-partials and inverse-FFT -> corr_mean[b, tau]
// scale = 1/(T * E)  (irfft 1/T times mean over H*HD = E channels)
// ---------------------------------------------------------------------------
__global__ __launch_bounds__(256) void reduce_ifft(
    const float* __restrict__ part, float* __restrict__ corr) {
  __shared__ float re[TT], im[TT];
  const int b = blockIdx.x;
  for (int f = threadIdx.x; f < TT; f += 256) {
    float sr = 0.f, si = 0.f;
    for (int g = 0; g < 64; ++g) {
      const float* p = &part[(((size_t)b * 64 + g) * TT + f) * 2];
      sr += p[0]; si += p[1];
    }
    const int r = __brev((unsigned)f) >> 21;
    re[r] = sr; im[r] = si;
  }
  __syncthreads();
  block_fft(re, im, +1.0f);                        // inverse (unnormalized)
  const float scale = 1.0f / ((float)TT * (float)EEC);
  for (int t = threadIdx.x; t < TT; t += 256)
    corr[b * TT + t] = re[t] * scale;
}

// ---------------------------------------------------------------------------
// Top-22 (ties -> lower index, matching lax.top_k) + softmax per batch.
// ---------------------------------------------------------------------------
__global__ __launch_bounds__(256) void topk_softmax(
    const float* __restrict__ corr, float* __restrict__ wout,
    int* __restrict__ iout) {
  __shared__ float cm[TT];
  __shared__ float rv[256];
  __shared__ int   ri[256];
  __shared__ float tv[TOPK];
  __shared__ int   ti[TOPK];
  const int b = blockIdx.x;
  for (int t = threadIdx.x; t < TT; t += 256) cm[t] = corr[b * TT + t];
  __syncthreads();
  for (int it = 0; it < TOPK; ++it) {
    float best = NEG_INF; int bi = TT;
    for (int t = threadIdx.x; t < TT; t += 256) {
      const float v = cm[t];
      if (v > best) { best = v; bi = t; }
    }
    rv[threadIdx.x] = best; ri[threadIdx.x] = bi;
    __syncthreads();
    for (int s = 128; s > 0; s >>= 1) {
      if (threadIdx.x < (unsigned)s) {
        const float ov = rv[threadIdx.x + s]; const int oi = ri[threadIdx.x + s];
        if (ov > rv[threadIdx.x] ||
            (ov == rv[threadIdx.x] && oi < ri[threadIdx.x])) {
          rv[threadIdx.x] = ov; ri[threadIdx.x] = oi;
        }
      }
      __syncthreads();
    }
    if (threadIdx.x == 0) { tv[it] = rv[0]; ti[it] = ri[0]; cm[ri[0]] = NEG_INF; }
    __syncthreads();
  }
  if (threadIdx.x == 0) {
    const float mx = tv[0];                        // first pick = global max
    float e[TOPK]; float sum = 0.f;
    for (int i = 0; i < TOPK; ++i) { e[i] = __expf(tv[i] - mx); sum += e[i]; }
    const float inv = 1.0f / sum;
    for (int i = 0; i < TOPK; ++i) {
      wout[b * TOPK + i] = e[i] * inv;
      iout[b * TOPK + i] = ti[i];
    }
  }
}

// ---------------------------------------------------------------------------
// agg[b,t,:] = sum_i w[b,i] * V[b, (t + d_i) mod T, :]   (bandwidth bound)
// ---------------------------------------------------------------------------
__global__ __launch_bounds__(256) void aggregate_delays(
    const float* __restrict__ V, const float* __restrict__ w22,
    const int* __restrict__ i22, float* __restrict__ outF) {
  __shared__ float w[TOPK];
  __shared__ int   d[TOPK];
  const int t = blockIdx.x, b = blockIdx.y;
  if (threadIdx.x < TOPK) {
    w[threadIdx.x] = w22[b * TOPK + threadIdx.x];
    d[threadIdx.x] = i22[b * TOPK + threadIdx.x];
  }
  __syncthreads();
  const int e = threadIdx.x << 2;                  // 256 thr * float4 = 1024
  float4 acc = make_float4(0.f, 0.f, 0.f, 0.f);
  for (int i = 0; i < TOPK; ++i) {
    int ts = t + d[i]; if (ts >= TT) ts -= TT;
    const float4 vv = *(const float4*)(V + ((size_t)(b * TT + ts) * EEC + e));
    acc.x += w[i] * vv.x; acc.y += w[i] * vv.y;
    acc.z += w[i] * vv.z; acc.w += w[i] * vv.w;
  }
  *(float4*)(outF + ((size_t)(b * TT + t) * EEC + e)) = acc;
}

// ===========================================================================
extern "C" void kernel_launch(void* const* d_in, const int* in_sizes, int n_in,
                              void* d_out, int out_size, void* d_ws, size_t ws_size,
                              hipStream_t stream) {
  (void)in_sizes; (void)n_in; (void)out_size; (void)ws_size;
  const float* X  = (const float*)d_in[0];
  const float* Wq = (const float*)d_in[1];
  const float* bq = (const float*)d_in[2];
  const float* Wk = (const float*)d_in[3];
  const float* bk = (const float*)d_in[4];
  const float* Wv = (const float*)d_in[5];
  const float* bv = (const float*)d_in[6];
  const float* Wo = (const float*)d_in[7];
  const float* bo = (const float*)d_in[8];

  const long long BTE = (long long)BB * TT * EEC;   // 33,554,432
  const long long EE2 = (long long)EEC * EEC;       // 1,048,576
  const int MR = BB * TT;                            // 32768 rows

  char* ws = (char*)d_ws;
  size_t off = 0;
  auto alloc = [&](size_t bytes) -> void* {
    void* p = ws + off;
    off = (off + bytes + 255) & ~(size_t)255;
    return p;
  };
  float*  qF   = (float*)alloc((size_t)BTE * 4);
  float*  kF   = (float*)alloc((size_t)BTE * 4);
  float*  vF   = (float*)alloc((size_t)BTE * 4);
  float*  aggF = (float*)alloc((size_t)BTE * 4);
  __bf16* xbf  = (__bf16*)alloc((size_t)BTE * 2);   // reused for agg (bf16)
  __bf16* wqbf = (__bf16*)alloc((size_t)EE2 * 2);
  __bf16* wkbf = (__bf16*)alloc((size_t)EE2 * 2);
  __bf16* wvbf = (__bf16*)alloc((size_t)EE2 * 2);
  __bf16* wobf = (__bf16*)alloc((size_t)EE2 * 2);
  float*  part = (float*)alloc((size_t)BB * 64 * TT * 2 * 4);
  float*  corr = (float*)alloc((size_t)BB * TT * 4);
  float*  topw = (float*)alloc((size_t)BB * TOPK * 4);
  int*    topi = (int*)alloc((size_t)BB * TOPK * 4);

  const int cbX = (int)((BTE + 255) / 256);
  const int cbW = (int)((EE2 + 255) / 256);
  const dim3 ggrid(EEC / BN, MR / BM);               // (16, 256)

  // 1) conversions
  convert_f32_to_bf16<<<cbX, 256, 0, stream>>>(X,  xbf,  BTE);
  convert_f32_to_bf16<<<cbW, 256, 0, stream>>>(Wq, wqbf, EE2);
  convert_f32_to_bf16<<<cbW, 256, 0, stream>>>(Wk, wkbf, EE2);
  convert_f32_to_bf16<<<cbW, 256, 0, stream>>>(Wv, wvbf, EE2);
  convert_f32_to_bf16<<<cbW, 256, 0, stream>>>(Wo, wobf, EE2);

  // 2) projections (WMMA bf16, f32 accum)
  gemm_bf16_wmma<<<ggrid, 256, 0, stream>>>(xbf, wqbf, bq, qF, MR, EEC, EEC);
  gemm_bf16_wmma<<<ggrid, 256, 0, stream>>>(xbf, wkbf, bk, kF, MR, EEC, EEC);
  gemm_bf16_wmma<<<ggrid, 256, 0, stream>>>(xbf, wvbf, bv, vF, MR, EEC, EEC);

  // 3) frequency-domain correlation, reduced over channels
  fft_corr_partial<<<dim3(64, BB), 256, 0, stream>>>(qF, kF, part);
  reduce_ifft<<<BB, 256, 0, stream>>>(part, corr);

  // 4) top-k delays + softmax weights
  topk_softmax<<<BB, 256, 0, stream>>>(corr, topw, topi);

  // 5) time-delay aggregation of V
  aggregate_delays<<<dim3(TT, BB), 256, 0, stream>>>(vF, topw, topi, aggF);

  // 6) output projection
  convert_f32_to_bf16<<<cbX, 256, 0, stream>>>(aggF, xbf, BTE);
  gemm_bf16_wmma<<<ggrid, 256, 0, stream>>>(xbf, wobf, bo, (float*)d_out,
                                            MR, EEC, EEC);
}
